// MultiHeadATT_86320252715606
// MI455X (gfx1250) — compile-verified, matrix operand order checked by reading, not run on hardware
//
#include <hip/hip_runtime.h>
#include <hip/hip_bf16.h>

// ---------------------------------------------------------------------------
// Fused MHA + out-proj + residual + LayerNorm for gfx1250 (MI455X).
// GEMMs via V_WMMA_F32_16X16X32_BF16; K/V + ctx tiles staged with
// GLOBAL_LOAD_ASYNC_TO_LDS_B128 (ASYNCcnt) when the toolchain exposes it.
// ---------------------------------------------------------------------------

typedef __attribute__((ext_vector_type(16))) __bf16 bf16x16;
typedef __attribute__((ext_vector_type(8)))  float  f32x8;
typedef int v4i __attribute__((vector_size(16)));   // matches builtin pointee

#define BATCH 4
#define SEQ   2048
#define DMODEL 512
#define NHEAD 8
#define DHEAD 64
#define NTOK  (BATCH * SEQ)     // 8192

#if __has_builtin(__builtin_amdgcn_global_load_async_to_lds_b128)
#define ASYNC_LDS 1
#else
#define ASYNC_LDS 0
#endif

// 16-byte async (or plain) global->LDS copy; per-lane addresses.
__device__ __forceinline__ void cp16_g2l(void* lds, const void* g) {
#if ASYNC_LDS
  auto gp = (__attribute__((address_space(1))) v4i*)(g);
  auto lp = (__attribute__((address_space(3))) v4i*)(lds);
  __builtin_amdgcn_global_load_async_to_lds_b128(gp, lp, 0, 0);
#else
  *(uint4*)lds = *(const uint4*)g;
#endif
}

__device__ __forceinline__ void wait_async0() {
#if ASYNC_LDS
#if __has_builtin(__builtin_amdgcn_s_wait_asynccnt)
  __builtin_amdgcn_s_wait_asynccnt(0);
#else
  asm volatile("s_wait_asynccnt 0" ::: "memory");
#endif
#endif
}

__device__ __forceinline__ f32x8 wmma_bf16(bf16x16 a, bf16x16 b, f32x8 c) {
  // D = A(16x32 bf16) * B(32x16 bf16) + C(16x16 f32)
  return __builtin_amdgcn_wmma_f32_16x16x32_bf16(
      /*neg_a=*/false, a, /*neg_b=*/false, b,
      /*c_mod=*/(short)0, c, /*reuse_a=*/false, /*reuse_b=*/false);
}

// ---- fragment loaders (per CDNA5 ISA VGPR layout tables) -------------------
// A 16x32 bf16 row-major at `t`, leading dim `ld` (elements, even).
__device__ __forceinline__ bf16x16 load_frag_a(const __bf16* t, int ld) {
  const int l = threadIdx.x & 31;
  const int half = l >> 4, m = l & 15;
  union { bf16x16 v; unsigned u[8]; } f;
  const __bf16* row = t + m * ld;
#pragma unroll
  for (int vg = 0; vg < 8; ++vg) {
    const int k = ((vg & 4) ? 16 : 0) + half * 8 + 2 * (vg & 3);
    f.u[vg] = *(const unsigned*)(row + k);
  }
  return f.v;
}

// B 32(K)x16(N) bf16, stored transposed: t[n*ld + k].
__device__ __forceinline__ bf16x16 load_frag_b_t(const __bf16* t, int ld) {
  const int l = threadIdx.x & 31;
  const int half = l >> 4, n = l & 15;
  union { bf16x16 v; unsigned u[8]; } f;
  const __bf16* row = t + n * ld;
#pragma unroll
  for (int vg = 0; vg < 8; ++vg) {
    const int k = ((vg & 4) ? 16 : 0) + half * 8 + 2 * (vg & 3);
    f.u[vg] = *(const unsigned*)(row + k);
  }
  return f.v;
}

// B 32(K)x16(N) bf16, stored row-major in K: t[k*ld + n].
__device__ __forceinline__ bf16x16 load_frag_b_rm(const __bf16* t, int ld) {
  const int l = threadIdx.x & 31;
  const int half = l >> 4, n = l & 15;
  bf16x16 v;
#pragma unroll
  for (int vg = 0; vg < 8; ++vg) {
    const int k = ((vg & 4) ? 16 : 0) + half * 8 + 2 * (vg & 3);
    v[2 * vg]     = t[k * ld + n];
    v[2 * vg + 1] = t[(k + 1) * ld + n];
  }
  return v;
}

__device__ __forceinline__ float rmax16(float v) {
#pragma unroll
  for (int m = 8; m >= 1; m >>= 1) v = fmaxf(v, __shfl_xor(v, m, 16));
  return v;
}
__device__ __forceinline__ float rsum16(float v) {
#pragma unroll
  for (int m = 8; m >= 1; m >>= 1) v += __shfl_xor(v, m, 16);
  return v;
}

// ---------------------------------------------------------------------------
// Kernel 1: projection  Y[b,h,s,dh] = sum_d X[b,s,d] * W[e,d] + bias[e]
// WG tile 128(M) x 64(N), 8 waves as 4x2; wave = 32x32 -> 4 WMMA / K-step.
// ---------------------------------------------------------------------------
#define PLD 40  // LDS leading dim: even and 16B-aligned rows (40*2 = 80 = 16*5)

__global__ void __launch_bounds__(256)
mha_proj_kernel(const float* __restrict__ X, const float* __restrict__ W,
                const float* __restrict__ bias, __bf16* __restrict__ out) {
  __shared__ __bf16 Xs[128 * PLD];
  __shared__ __bf16 Ws[64 * PLD];

  const int tid  = threadIdx.x;
  const int wave = tid >> 5;
  const int lane = tid & 31;
  const int wm = wave >> 1;          // 0..3 -> 32-row M sub-tile
  const int wn = wave & 1;           // 0..1 -> 32-col N sub-tile
  const int m0 = blockIdx.x * 128;
  const int n0 = blockIdx.y * 64;

  f32x8 acc[2][2] = {};

  for (int kk = 0; kk < DMODEL; kk += 32) {
#pragma unroll
    for (int i = 0; i < 16; ++i) {          // X tile: 128x32 f32 -> bf16
      const int e = tid + 256 * i;
      const int r = e >> 5, c = e & 31;
      Xs[r * PLD + c] = (__bf16)X[(m0 + r) * DMODEL + kk + c];
    }
#pragma unroll
    for (int i = 0; i < 8; ++i) {           // W tile: 64x32 f32 -> bf16
      const int e = tid + 256 * i;
      const int r = e >> 5, c = e & 31;
      Ws[r * PLD + c] = (__bf16)W[(n0 + r) * DMODEL + kk + c];
    }
    __syncthreads();

    bf16x16 a[2], b[2];
#pragma unroll
    for (int rt = 0; rt < 2; ++rt)
      a[rt] = load_frag_a(Xs + (32 * wm + 16 * rt) * PLD, PLD);
#pragma unroll
    for (int f = 0; f < 2; ++f)
      b[f] = load_frag_b_t(Ws + (32 * wn + 16 * f) * PLD, PLD);
#pragma unroll
    for (int rt = 0; rt < 2; ++rt)
#pragma unroll
      for (int f = 0; f < 2; ++f)
        acc[rt][f] = wmma_bf16(a[rt], b[f], acc[rt][f]);
    __syncthreads();
  }

  // epilogue: +bias, store bf16 in [B, H, S, DH] layout
  const int half = lane >> 4, nl = lane & 15;
#pragma unroll
  for (int f = 0; f < 2; ++f) {
    const int colg = n0 + 32 * wn + 16 * f + nl;   // e index
    const int h = colg >> 6, dh = colg & 63;
    const float bv = bias[colg];
#pragma unroll
    for (int rt = 0; rt < 2; ++rt)
#pragma unroll
      for (int r = 0; r < 8; ++r) {
        const int m = m0 + 32 * wm + 16 * rt + half * 8 + r;
        const int b_ = m >> 11, s = m & (SEQ - 1);
        out[(((b_ * NHEAD + h) * SEQ) + s) * DHEAD + dh] =
            (__bf16)(acc[rt][f][r] + bv);
      }
  }
}

// ---------------------------------------------------------------------------
// Kernel 2: flash attention per (b,h).  4 waves/WG, wave = 32 query rows.
// Per 32-kv step per wave: 8 score WMMAs + online softmax + 8 ctx WMMAs.
// scores = q*k^T / sqrt(H)   (reference quirk: sqrt(n_head), not sqrt(d_k))
// ---------------------------------------------------------------------------
#define KLD 72   // K/V LDS rows: 72*2 = 144 B = 16B-aligned
#define PLD2 34  // P relayout tile leading dim (even -> 4B-aligned pairs)

__global__ void __launch_bounds__(128)
mha_attn_kernel(const __bf16* __restrict__ qp, const __bf16* __restrict__ kp,
                const __bf16* __restrict__ vp, __bf16* __restrict__ ctx) {
  __shared__ __bf16 Kt[32 * KLD];
  __shared__ __bf16 Vt[32 * KLD];
  __shared__ __bf16 Pt[4][32 * PLD2];

  const int tid  = threadIdx.x;
  const int wave = tid >> 5;
  const int lane = tid & 31;
  const int half = lane >> 4, nl = lane & 15;
  const int bh = blockIdx.y;                    // b*8 + h
  const int q0 = blockIdx.x * 128 + wave * 32;  // 32 query rows per wave
  const float scale = 0.35355339059f;           // 1/sqrt(8)

  // q fragments straight from global (bf16, [B,H,S,DH]; rows are 128 B)
  bf16x16 qa[2][2];
#pragma unroll
  for (int rt = 0; rt < 2; ++rt) {
    const __bf16* qb = qp + (bh * SEQ + q0 + 16 * rt) * DHEAD;
    qa[rt][0] = load_frag_a(qb, DHEAD);        // dh 0..31
    qa[rt][1] = load_frag_a(qb + 32, DHEAD);   // dh 32..63
  }

  float mrow[2][8], lrow[2][8];
#pragma unroll
  for (int rt = 0; rt < 2; ++rt)
#pragma unroll
    for (int r = 0; r < 8; ++r) { mrow[rt][r] = -1e30f; lrow[rt][r] = 0.f; }
  f32x8 O[2][4] = {};

  const __bf16* kbh = kp + bh * SEQ * DHEAD;
  const __bf16* vbh = vp + bh * SEQ * DHEAD;

  for (int kv0 = 0; kv0 < SEQ; kv0 += 32) {
    // async-stage K/V 32x64 bf16 tiles: 256 16B-chunks each, 2/thread/tile
#pragma unroll
    for (int i = 0; i < 2; ++i) {
      const int c = tid + 128 * i;
      const int r = c >> 3, c8 = (c & 7) * 8;
      cp16_g2l(&Kt[r * KLD + c8], &kbh[(kv0 + r) * DHEAD + c8]);
      cp16_g2l(&Vt[r * KLD + c8], &vbh[(kv0 + r) * DHEAD + c8]);
    }
    if (kv0 + 32 < SEQ) {                       // pull next tiles toward L2
      __builtin_prefetch(&kbh[(kv0 + 32) * DHEAD + (tid & 31) * 4], 0, 1);
      __builtin_prefetch(&vbh[(kv0 + 32) * DHEAD + (tid & 31) * 4], 0, 1);
    }
    wait_async0();
    __syncthreads();

    // score fragments: kb[j][kc] covers kv cols 16j, dh chunk 32*kc
    bf16x16 kb[2][2];
#pragma unroll
    for (int j = 0; j < 2; ++j) {
      kb[j][0] = load_frag_b_t(&Kt[(16 * j) * KLD + 0], KLD);
      kb[j][1] = load_frag_b_t(&Kt[(16 * j) * KLD + 32], KLD);
    }
    f32x8 s[2][2];
#pragma unroll
    for (int rt = 0; rt < 2; ++rt)
#pragma unroll
      for (int j = 0; j < 2; ++j) {
        f32x8 z = {};
        z = wmma_bf16(qa[rt][0], kb[j][0], z);
        s[rt][j] = wmma_bf16(qa[rt][1], kb[j][1], z);
      }

    // online softmax (rows live in half-wave groups of 16 lanes)
    __bf16* pw = Pt[wave];
#pragma unroll
    for (int rt = 0; rt < 2; ++rt)
#pragma unroll
      for (int r = 0; r < 8; ++r) {
        const float a0 = s[rt][0][r] * scale, a1 = s[rt][1][r] * scale;
        const float bm = rmax16(fmaxf(a0, a1));
        const float mn = fmaxf(mrow[rt][r], bm);
        const float al = __expf(mrow[rt][r] - mn);
        const float e0 = __expf(a0 - mn);
        const float e1 = __expf(a1 - mn);
        const float rs = rsum16(e0 + e1);
        lrow[rt][r] = lrow[rt][r] * al + rs;
        mrow[rt][r] = mn;
#pragma unroll
        for (int j = 0; j < 4; ++j) O[rt][j][r] *= al;
        const int m = 16 * rt + half * 8 + r;
        pw[m * PLD2 + nl]      = (__bf16)e0;
        pw[m * PLD2 + nl + 16] = (__bf16)e1;
      }
    __builtin_amdgcn_wave_barrier();
    asm volatile("s_wait_dscnt 0" ::: "memory");  // CDNA5 split LDS counter

    // O += P(32x32) * V(32x64): shared V fragments across both row tiles
    bf16x16 vb[4];
#pragma unroll
    for (int j = 0; j < 4; ++j) vb[j] = load_frag_b_rm(&Vt[16 * j], KLD);
#pragma unroll
    for (int rt = 0; rt < 2; ++rt) {
      const bf16x16 pa = load_frag_a(pw + (16 * rt) * PLD2, PLD2);
#pragma unroll
      for (int j = 0; j < 4; ++j) O[rt][j] = wmma_bf16(pa, vb[j], O[rt][j]);
    }
    __syncthreads();
  }

  // normalize; store ctx as bf16 in [B, S, H*DH] layout for out-proj GEMM
  const int b = bh >> 3, h = bh & 7;
#pragma unroll
  for (int rt = 0; rt < 2; ++rt)
#pragma unroll
    for (int r = 0; r < 8; ++r) {
      const int sg = q0 + 16 * rt + half * 8 + r;
      const float inv = 1.f / lrow[rt][r];
      __bf16* dst = ctx + (b * SEQ + sg) * DMODEL + h * DHEAD + nl;
      dst[0]  = (__bf16)(O[rt][0][r] * inv);
      dst[16] = (__bf16)(O[rt][1][r] * inv);
      dst[32] = (__bf16)(O[rt][2][r] * inv);
      dst[48] = (__bf16)(O[rt][3][r] * inv);
    }
}

// ---------------------------------------------------------------------------
// Kernel 3: out-proj + bias + residual -> f32 scratch
//   x[b,s,d] = sum_e ctx[b,s,e] * Wo[d,e] + bo[d] + Q[b,s,d]
// Same 128x64 tiling; ctx tile is already bf16 -> async B128 stage.
// ---------------------------------------------------------------------------
__global__ void __launch_bounds__(256)
mha_outproj_kernel(const __bf16* __restrict__ ctx, const float* __restrict__ Wo,
                   const float* __restrict__ bo, const float* __restrict__ Qres,
                   float* __restrict__ xres) {
  __shared__ __bf16 Xs[128 * PLD];
  __shared__ __bf16 Ws[64 * PLD];

  const int tid  = threadIdx.x;
  const int wave = tid >> 5;
  const int lane = tid & 31;
  const int wm = wave >> 1, wn = wave & 1;
  const int m0 = blockIdx.x * 128;
  const int n0 = blockIdx.y * 64;

  f32x8 acc[2][2] = {};

  for (int kk = 0; kk < DMODEL; kk += 32) {
#pragma unroll
    for (int i = 0; i < 2; ++i) {           // ctx 128x32 bf16: async 16B chunks
      const int c = tid + 256 * i;          // 512 chunks
      const int r = c >> 2, c8 = (c & 3) * 8;
      cp16_g2l(&Xs[r * PLD + c8], &ctx[(m0 + r) * DMODEL + kk + c8]);
    }
#pragma unroll
    for (int i = 0; i < 8; ++i) {           // Wo: f32 -> bf16
      const int e = tid + 256 * i;
      const int r = e >> 5, c = e & 31;
      Ws[r * PLD + c] = (__bf16)Wo[(n0 + r) * DMODEL + kk + c];
    }
    wait_async0();
    __syncthreads();

    bf16x16 a[2], b[2];
#pragma unroll
    for (int rt = 0; rt < 2; ++rt)
      a[rt] = load_frag_a(Xs + (32 * wm + 16 * rt) * PLD, PLD);
#pragma unroll
    for (int f = 0; f < 2; ++f)
      b[f] = load_frag_b_t(Ws + (32 * wn + 16 * f) * PLD, PLD);
#pragma unroll
    for (int rt = 0; rt < 2; ++rt)
#pragma unroll
      for (int f = 0; f < 2; ++f)
        acc[rt][f] = wmma_bf16(a[rt], b[f], acc[rt][f]);
    __syncthreads();
  }

  const int half = lane >> 4, nl = lane & 15;
#pragma unroll
  for (int f = 0; f < 2; ++f) {
    const int colg = n0 + 32 * wn + 16 * f + nl;    // d index
    const float bv = bo[colg];
#pragma unroll
    for (int rt = 0; rt < 2; ++rt)
#pragma unroll
      for (int r = 0; r < 8; ++r) {
        const int m = m0 + 32 * wm + 16 * rt + half * 8 + r;
        xres[m * DMODEL + colg] = acc[rt][f][r] + bv + Qres[m * DMODEL + colg];
      }
  }
}

// ---------------------------------------------------------------------------
// Kernel 4: LayerNorm over D=512, one wave per row.
// ---------------------------------------------------------------------------
__global__ void __launch_bounds__(256)
mha_ln_kernel(const float* __restrict__ x, const float* __restrict__ gamma,
              const float* __restrict__ beta, float* __restrict__ out) {
  const int lane = threadIdx.x & 31;
  const int row = blockIdx.x * 8 + (threadIdx.x >> 5);
  const float* xr = x + row * DMODEL;

  float vals[16], sum = 0.f, sq = 0.f;
#pragma unroll
  for (int i = 0; i < 16; ++i) {
    const float v = xr[lane + 32 * i];
    vals[i] = v; sum += v; sq += v * v;
  }
#pragma unroll
  for (int m = 16; m >= 1; m >>= 1) {
    sum += __shfl_xor(sum, m, 32);
    sq  += __shfl_xor(sq,  m, 32);
  }
  const float mean = sum * (1.f / DMODEL);
  const float var  = sq * (1.f / DMODEL) - mean * mean;
  const float rstd = rsqrtf(var + 1e-5f);
#pragma unroll
  for (int i = 0; i < 16; ++i) {
    const int c = lane + 32 * i;
    out[row * DMODEL + c] = (vals[i] - mean) * rstd * gamma[c] + beta[c];
  }
}

// ---------------------------------------------------------------------------
extern "C" void kernel_launch(void* const* d_in, const int* in_sizes, int n_in,
                              void* d_out, int out_size, void* d_ws, size_t ws_size,
                              hipStream_t stream) {
  (void)in_sizes; (void)n_in; (void)out_size; (void)ws_size;
  const float* Q     = (const float*)d_in[0];
  const float* K     = (const float*)d_in[1];
  const float* V     = (const float*)d_in[2];
  const float* Wq    = (const float*)d_in[3];
  const float* bq    = (const float*)d_in[4];
  const float* Wk    = (const float*)d_in[5];
  const float* bk    = (const float*)d_in[6];
  const float* Wv    = (const float*)d_in[7];
  const float* bv    = (const float*)d_in[8];
  const float* Wo    = (const float*)d_in[9];
  const float* bo    = (const float*)d_in[10];
  const float* gamma = (const float*)d_in[11];
  const float* beta  = (const float*)d_in[12];
  float* out = (float*)d_out;

  // workspace: 4x bf16 [8192,512] + 1x f32 [8192,512]  (48 MiB total)
  char* ws = (char*)d_ws;
  const size_t bf16Mat = (size_t)NTOK * DMODEL * sizeof(__bf16);
  __bf16* qp  = (__bf16*)(ws);
  __bf16* kp  = (__bf16*)(ws + 1 * bf16Mat);
  __bf16* vp  = (__bf16*)(ws + 2 * bf16Mat);
  __bf16* ctx = (__bf16*)(ws + 3 * bf16Mat);
  float*  xr  = (float*) (ws + 4 * bf16Mat);

  const dim3 gemmGrid(NTOK / 128, DMODEL / 64);   // 64 x 8
  mha_proj_kernel<<<gemmGrid, 256, 0, stream>>>(Q, Wq, bq, qp);
  mha_proj_kernel<<<gemmGrid, 256, 0, stream>>>(K, Wk, bk, kp);
  mha_proj_kernel<<<gemmGrid, 256, 0, stream>>>(V, Wv, bv, vp);

  const dim3 attnGrid(SEQ / 128, BATCH * NHEAD);  // 16 x 32
  mha_attn_kernel<<<attnGrid, 128, 0, stream>>>(qp, kp, vp, ctx);

  mha_outproj_kernel<<<gemmGrid, 256, 0, stream>>>(ctx, Wo, bo, Q, xr);

  mha_ln_kernel<<<NTOK / 8, 256, 0, stream>>>(xr, gamma, beta, out);
}
